// BidirectionalDecoder_44994077392980
// MI455X (gfx1250) — matrix operand order repaired and checked by reading, not compiled, not run on hardware
//
#include <hip/hip_runtime.h>

// ---------------------------------------------------------------------------
// BidirectionalDecoder for MI455X (gfx1250, wave32, WMMA bf16 + TDM async DMA)
// B=64, S=2048, ENC=1024, U=512, E=256, V=50000, IN_DIM=1280
// ---------------------------------------------------------------------------

typedef __attribute__((ext_vector_type(16))) __bf16 v16bf;
typedef __attribute__((ext_vector_type(8)))  float  v8f;
typedef __attribute__((ext_vector_type(4)))  unsigned int v4u;
typedef __attribute__((ext_vector_type(4)))  int    v4i;
typedef __attribute__((ext_vector_type(8)))  int    v8i;

__device__ __forceinline__ unsigned short f32_to_bf16(float f) {
  union { float f; unsigned u; } c; c.f = f;
  unsigned r = c.u + 0x7FFFu + ((c.u >> 16) & 1u);   // round-to-nearest-even
  return (unsigned short)(r >> 16);
}

// LDS byte offset of a __shared__ object (generic LDS addr: offset in [31:0]).
__device__ __forceinline__ unsigned lds_off_of(const void* p) {
  return (unsigned)(unsigned long long)p;
}

// ---------------------------------------------------------------------------
// Tensor Data Mover: async 2D fp32 tile load Global -> LDS.
// D# per CDNA5 ISA: group0 = {count, lds_addr,
// global_addr, type=2}; group1 = {data_size=4B, dims, tile dims, stride};
// groups 2/3 zero (2D tensor). Tracked with TENSORcnt.
// ---------------------------------------------------------------------------
__device__ __forceinline__ void tdm_load_2d_f32(unsigned lds_byte_off,
                                                const float* gptr,
                                                unsigned tile_w,   // elems (X)
                                                unsigned tile_h,   // rows  (Y)
                                                unsigned row_stride /* elems */) {
  unsigned long long ga = (unsigned long long)gptr;
  v4u g0;
  g0[0] = 1u;                                        // count=1, user descriptor
  g0[1] = lds_byte_off;                              // LDS dest (bytes)
  g0[2] = (unsigned)(ga & 0xffffffffu);              // global_addr[31:0]
  g0[3] = (unsigned)((ga >> 32) & 0x1ffffffu)        // global_addr[56:32]
        | (2u << 30);                                // type=2 ("image")
  v8i g1;
  g1[0] = (int)(2u << 16);                           // data_size=2 -> 4 bytes
  g1[1] = (int)((tile_w & 0xffffu) << 16);           // tensor_dim0[15:0]
  g1[2] = (int)(((tile_w >> 16) & 0xffffu) | ((tile_h & 0xffffu) << 16));
  g1[3] = (int)(((tile_h >> 16) & 0xffffu) | ((tile_w & 0xffffu) << 16)); // tile_dim0
  g1[4] = (int)(tile_h & 0xffffu);                   // tile_dim1 (tile_dim2=0)
  g1[5] = (int)row_stride;                           // tensor_dim0_stride[31:0]
  g1[6] = 0;                                         // stride0_hi | stride1_lo
  g1[7] = 0;
  v4i z4 = {0, 0, 0, 0};
#if defined(__clang_major__) && (__clang_major__ >= 23)
  v8i z8 = {0, 0, 0, 0, 0, 0, 0, 0};
  __builtin_amdgcn_tensor_load_to_lds(g0, g1, z4, z4, z8, 0);
#else
  __builtin_amdgcn_tensor_load_to_lds(g0, g1, z4, z4, 0);
#endif
}

// A fragment (16x32 bf16) from LDS tile with row stride 32 halves.
// lanes 0-15 -> M=lane, K=0..7 & 16..23 ; lanes 16-31 -> K=8..15 & 24..31
__device__ __forceinline__ v16bf load_a_frag(const unsigned short* lds_a, int lane) {
  const int row = lane & 15;
  const int kh  = (lane >> 4) * 8;
  union { uint4 q[2]; v16bf v; } r;
  r.q[0] = *(const uint4*)(lds_a + row * 32 + kh);
  r.q[1] = *(const uint4*)(lds_a + row * 32 + kh + 16);
  return r.v;
}

// B fragment (32x16 bf16) from packed col-major bf16 weights W[n*Kdim + k].
__device__ __forceinline__ v16bf load_b_frag(const unsigned short* Wt, int Kdim,
                                             int colbase, int k0, int lane) {
  const unsigned short* p = Wt + (size_t)(colbase + (lane & 15)) * Kdim
                               + k0 + (lane >> 4) * 8;
  union { uint4 q[2]; v16bf v; } r;
  r.q[0] = *(const uint4*)(p);
  r.q[1] = *(const uint4*)(p + 16);
  return r.v;
}

__device__ __forceinline__ v8f wmma_bf16(v16bf a, v16bf b, v8f c) {
  return __builtin_amdgcn_wmma_f32_16x16x32_bf16(false, a, false, b,
                                                 (short)0, c, false, false);
}

__device__ __forceinline__ float sigmoidf(float x) { return 1.f / (1.f + expf(-x)); }

// ---------------------------------------------------------------------------
// Prep: pack W1 (1024x512 f32 row-major) -> W1t (512x1024 bf16, n-major)
// ---------------------------------------------------------------------------
__global__ void pack_w1_kernel(const float* __restrict__ W1,
                               unsigned short* __restrict__ W1t) {
  int idx = blockIdx.x * 256 + threadIdx.x;           // 512*1024
  int n = idx >> 10, k = idx & 1023;
  W1t[idx] = f32_to_bf16(W1[(size_t)k * 512 + n]);
}

// Pack Kf/Kb (1280x2048 f32 row-major) -> (2048x1280 bf16, n-major)
__global__ void pack_k_kernel(const float* __restrict__ Kf,
                              const float* __restrict__ Kb,
                              unsigned short* __restrict__ KfT,
                              unsigned short* __restrict__ KbT) {
  int idx = blockIdx.x * 256 + threadIdx.x;           // 2048*1280
  const float* src = blockIdx.y ? Kb : Kf;
  unsigned short* dst = blockIdx.y ? KbT : KfT;
  int n = idx / 1280, k = idx - n * 1280;
  dst[idx] = f32_to_bf16(src[(size_t)k * 2048 + n]);
}

// ---------------------------------------------------------------------------
// hidden_proj = c_fwd@W2 + b2 + c_bwd@W3 + b3   (64x512, tiny)
// ---------------------------------------------------------------------------
__global__ __launch_bounds__(256) void hproj_kernel(
    const float* __restrict__ c_fwd, const float* __restrict__ c_bwd,
    const float* __restrict__ W2, const float* __restrict__ b2,
    const float* __restrict__ W3, const float* __restrict__ b3,
    float* __restrict__ hp) {
  __shared__ float cf[64 * 64], cb[64 * 64];
  const int n = blockIdx.x * 256 + threadIdx.x;       // 512 cols
  float acc[64];
#pragma unroll
  for (int b = 0; b < 64; ++b) acc[b] = b2[n] + b3[n];
  for (int kc = 0; kc < 512; kc += 64) {
    __syncthreads();
    for (int i = threadIdx.x; i < 64 * 64; i += 256) {
      int bb = i >> 6, kk = i & 63;
      cf[i] = c_fwd[bb * 512 + kc + kk];
      cb[i] = c_bwd[bb * 512 + kc + kk];
    }
    __syncthreads();
    for (int kk = 0; kk < 64; ++kk) {
      float w2 = W2[(size_t)(kc + kk) * 512 + n];
      float w3 = W3[(size_t)(kc + kk) * 512 + n];
#pragma unroll
      for (int b = 0; b < 64; ++b)
        acc[b] += cf[b * 64 + kk] * w2 + cb[b * 64 + kk] * w3;
    }
  }
#pragma unroll
  for (int b = 0; b < 64; ++b) hp[b * 512 + n] = acc[b];
}

// ---------------------------------------------------------------------------
// score = tanh(enc @ W1 + W1_b + hproj[b]) @ V_w + V_b   (fused, 137 GFLOP)
// Block: 256 thr = 8 waves, tile M=64 x N=512 (full U), K stepped by 32.
// enc tiles (64x32 fp32) DMA'd Global->LDS by the TDM, double-buffered,
// synchronized with TENSORcnt; converted once to a bf16 LDS tile; 16 WMMAs
// per K-step per wave. Fused tanh + V_w reduction epilogue.
// ---------------------------------------------------------------------------
__global__ __launch_bounds__(256) void score_kernel(
    const float* __restrict__ enc, const unsigned short* __restrict__ W1t,
    const float* __restrict__ W1_b, const float* __restrict__ hproj,
    const float* __restrict__ V_w, const float* __restrict__ V_b,
    float* __restrict__ score) {
  __shared__ __align__(16) float lds_f[2][64 * 32];        // TDM fp32 buffers
  __shared__ __align__(16) unsigned short lds_a[64 * 32];  // bf16 A tile
  __shared__ float lds_red[64];
  const int tid = threadIdx.x, lane = tid & 31, wave = tid >> 5;
  const int row_base = blockIdx.x * 64;               // flat row in [0, 131072)
  const int b = row_base >> 11;                       // batch (2048 rows/batch)
  const int ncb = wave * 64;
  const float* enc_base = enc + (size_t)row_base * 1024;

  if (wave == 0) {                                    // TDM ignores EXEC; one wave issues
    tdm_load_2d_f32(lds_off_of(lds_f[0]), enc_base + 0,  32, 64, 1024);
    tdm_load_2d_f32(lds_off_of(lds_f[1]), enc_base + 32, 32, 64, 1024);
  }

  v8f acc[4][4] = {};                                 // [mtile][ntile]
  for (int it = 0; it < 32; ++it) {
    if (wave == 0) __builtin_amdgcn_s_wait_tensorcnt((short)1); // oldest done
    __syncthreads();                                  // publish TDM tile to all waves
    // fp32 -> bf16 conversion pass (2048 elems, 8 contiguous per thread)
    {
      const float* fb = lds_f[it & 1];
      const int i0 = tid * 8;
      float4 x0 = *(const float4*)(fb + i0);
      float4 x1 = *(const float4*)(fb + i0 + 4);
      unsigned p0 = (unsigned)f32_to_bf16(x0.x) | ((unsigned)f32_to_bf16(x0.y) << 16);
      unsigned p1 = (unsigned)f32_to_bf16(x0.z) | ((unsigned)f32_to_bf16(x0.w) << 16);
      unsigned p2 = (unsigned)f32_to_bf16(x1.x) | ((unsigned)f32_to_bf16(x1.y) << 16);
      unsigned p3 = (unsigned)f32_to_bf16(x1.z) | ((unsigned)f32_to_bf16(x1.w) << 16);
      uint4 pk; pk.x = p0; pk.y = p1; pk.z = p2; pk.w = p3;
      *(uint4*)(lds_a + i0) = pk;
    }
    __syncthreads();                                  // bf16 tile ready; fp32 buf free
    if (wave == 0 && it + 2 < 32)                     // refill consumed buffer (async)
      tdm_load_2d_f32(lds_off_of(lds_f[it & 1]), enc_base + (it + 2) * 32,
                      32, 64, 1024);
    const int k0 = it * 32;
    v16bf afr[4];
#pragma unroll
    for (int m = 0; m < 4; ++m) afr[m] = load_a_frag(lds_a + m * 16 * 32, lane);
#pragma unroll
    for (int t = 0; t < 4; ++t) {
      v16bf bfr = load_b_frag(W1t, 1024, ncb + t * 16, k0, lane);
#pragma unroll
      for (int m = 0; m < 4; ++m) acc[m][t] = wmma_bf16(afr[m], bfr, acc[m][t]);
    }
  }
  // Fused epilogue: tanh(+bias) then dot with V_w, reduce across columns.
  if (tid < 64) lds_red[tid] = 0.f;
  __syncthreads();
  const int rowoff = (lane >> 4) * 8;                 // C layout: lanes>=16 -> M+8
#pragma unroll
  for (int m = 0; m < 4; ++m) {
    float rowsum[8];
#pragma unroll
    for (int r = 0; r < 8; ++r) rowsum[r] = 0.f;
#pragma unroll
    for (int t = 0; t < 4; ++t) {
      int col = ncb + t * 16 + (lane & 15);
      float bias = W1_b[col] + hproj[b * 512 + col];
      float vw = V_w[col];
#pragma unroll
      for (int r = 0; r < 8; ++r)
        rowsum[r] += tanhf(acc[m][t][r] + bias) * vw;
    }
#pragma unroll
    for (int r = 0; r < 8; ++r)
      atomicAdd(&lds_red[m * 16 + rowoff + r], rowsum[r]);
  }
  __syncthreads();
  if (tid < 64) score[row_base + tid] = lds_red[tid] + V_b[0];
}

// ---------------------------------------------------------------------------
// softmax over S=2048 per batch row
// ---------------------------------------------------------------------------
__global__ __launch_bounds__(256) void softmax_kernel(
    const float* __restrict__ score, float* __restrict__ attn) {
  __shared__ float red[256];
  const int b = blockIdx.x, tid = threadIdx.x;
  const float* s = score + b * 2048;
  float m = -3.0e38f;
  for (int i = tid; i < 2048; i += 256) m = fmaxf(m, s[i]);
  red[tid] = m; __syncthreads();
  for (int off = 128; off > 0; off >>= 1) {
    if (tid < off) red[tid] = fmaxf(red[tid], red[tid + off]);
    __syncthreads();
  }
  m = red[0]; __syncthreads();
  float sum = 0.f;
  for (int i = tid; i < 2048; i += 256) sum += expf(s[i] - m);
  red[tid] = sum; __syncthreads();
  for (int off = 128; off > 0; off >>= 1) {
    if (tid < off) red[tid] += red[tid + off];
    __syncthreads();
  }
  float inv = 1.f / red[0];
  for (int i = tid; i < 2048; i += 256) attn[b * 2048 + i] = expf(s[i] - m) * inv;
}

// ---------------------------------------------------------------------------
// ctx[b, e] = sum_s attn[b,s] * enc[b,s,e]  -> xin[:, 0:1024] (HBM-bound)
// ---------------------------------------------------------------------------
__global__ __launch_bounds__(256) void ctx_kernel(
    const float* __restrict__ attn, const float* __restrict__ enc,
    float* __restrict__ xin) {
  const int b = blockIdx.x;
  const int e = blockIdx.y * 256 + threadIdx.x;       // 0..1023
  const float* a = attn + b * 2048;
  const float* E = enc + (size_t)b * 2048 * 1024 + e;
  float acc = 0.f;
  for (int s = 0; s < 2048; ++s) acc += a[s] * E[(size_t)s * 1024];
  xin[b * 1280 + e] = acc;
}

// xin[:, 1024:1280] = emb[x[b]]
__global__ void emb_kernel(const int* __restrict__ x,
                           const float* __restrict__ emb,
                           float* __restrict__ xin) {
  const int b = blockIdx.x, e = threadIdx.x;          // 256 threads
  xin[b * 1280 + 1024 + e] = emb[(size_t)x[b] * 256 + e];
}

// ---------------------------------------------------------------------------
// LSTM gate GEMM: z[dir] = xin @ K[dir] + bias[dir]   (64x1280)@(1280x2048)
// ---------------------------------------------------------------------------
__global__ __launch_bounds__(256) void lstm_gemm_kernel(
    const float* __restrict__ xin,
    const unsigned short* __restrict__ KfT, const unsigned short* __restrict__ KbT,
    const float* __restrict__ bf_, const float* __restrict__ bb_,
    float* __restrict__ zbuf) {
  __shared__ __align__(16) unsigned short lds_a[64 * 32];
  const int tid = threadIdx.x, lane = tid & 31, wave = tid >> 5;
  const int dir = blockIdx.y;
  const unsigned short* Kt = dir ? KbT : KfT;
  const float* bias = dir ? bb_ : bf_;
  const int ncol = blockIdx.x * 64 + (wave & 3) * 16; // n-tile base col
  const int mbase = (wave >> 2) * 32;                 // rows [mbase, mbase+32)

  v8f acc0 = {}, acc1 = {};
  for (int k0 = 0; k0 < 1280; k0 += 32) {
    __syncthreads();
    for (int i = tid; i < 64 * 32; i += 256) {
      int r = i >> 5, kk = i & 31;
      lds_a[i] = f32_to_bf16(xin[r * 1280 + k0 + kk]);
    }
    __syncthreads();
    v16bf a0 = load_a_frag(lds_a + mbase * 32, lane);
    v16bf a1 = load_a_frag(lds_a + (mbase + 16) * 32, lane);
    v16bf bfr = load_b_frag(Kt, 1280, ncol, k0, lane);
    acc0 = wmma_bf16(a0, bfr, acc0);
    acc1 = wmma_bf16(a1, bfr, acc1);
  }
  const int col = ncol + (lane & 15);
  const float bv = bias[col];
  const int rowoff = (lane >> 4) * 8;
  float* zd = zbuf + (size_t)dir * 64 * 2048;
#pragma unroll
  for (int r = 0; r < 8; ++r) {
    zd[(size_t)(mbase + rowoff + r) * 2048 + col]      = acc0[r] + bv;
    zd[(size_t)(mbase + 16 + rowoff + r) * 2048 + col] = acc1[r] + bv;
  }
}

// ---------------------------------------------------------------------------
// Gates: c = sigmoid(i)*tanh(g); h = sigmoid(o)*tanh(c)  (zero initial state)
// ---------------------------------------------------------------------------
__global__ void gates_kernel(const float* __restrict__ zbuf,
                             float* __restrict__ outbuf,
                             float* __restrict__ dout) {
  const int dir = blockIdx.y;
  const int idx = blockIdx.x * 256 + threadIdx.x;     // 0..32767
  const int b = idx >> 9, u = idx & 511;
  const float* z = zbuf + (size_t)dir * 64 * 2048 + (size_t)b * 2048;
  float iv = z[u], gv = z[1024 + u], ov = z[1536 + u];
  float c = sigmoidf(iv) * tanhf(gv);
  float h = sigmoidf(ov) * tanhf(c);
  outbuf[b * 1024 + dir * 512 + u] = h;
  dout[3200000 + dir * 65536 + b * 512 + u] = h;      // hf | hb
  dout[3232768 + dir * 65536 + b * 512 + u] = c;      // cf | cb
}

// ---------------------------------------------------------------------------
// logits = out @ fc_w + fc_b  (64x1024)@(1024x50000); fc_w streamed ONCE.
// ---------------------------------------------------------------------------
__global__ __launch_bounds__(256) void logits_kernel(
    const float* __restrict__ outb, const float* __restrict__ fcw,
    const float* __restrict__ fcb, float* __restrict__ logits) {
  __shared__ float so[64 * 32];
  const int v = blockIdx.x * 256 + threadIdx.x;
  const bool ok = (v < 50000);
  float acc[64];
  float bias = ok ? fcb[v] : 0.f;
#pragma unroll
  for (int b = 0; b < 64; ++b) acc[b] = bias;
  for (int kc = 0; kc < 1024; kc += 32) {
    __syncthreads();
    for (int i = threadIdx.x; i < 64 * 32; i += 256) {
      int bb = i >> 5, kk = i & 31;
      so[i] = outb[bb * 1024 + kc + kk];
    }
    __syncthreads();
    if (ok && kc + 32 < 1024)                          // global_prefetch next slab
      __builtin_prefetch(&fcw[(size_t)(kc + 32) * 50000 + v], 0, 1);
    for (int kk = 0; kk < 32; ++kk) {
      float w = ok ? fcw[(size_t)(kc + kk) * 50000 + v] : 0.f;
#pragma unroll
      for (int b = 0; b < 64; ++b) acc[b] += so[b * 32 + kk] * w;
    }
  }
  if (ok) {
#pragma unroll
    for (int b = 0; b < 64; ++b) logits[(size_t)b * 50000 + v] = acc[b];
  }
}

// ---------------------------------------------------------------------------
// Workspace layout (bytes)
// ---------------------------------------------------------------------------
static constexpr size_t OFF_W1T = 0;
static constexpr size_t OFF_KFT = OFF_W1T + (size_t)512 * 1024 * 2;   // 1.0 MB
static constexpr size_t OFF_KBT = OFF_KFT + (size_t)2048 * 1280 * 2;  // 5.2 MB
static constexpr size_t OFF_HP  = OFF_KBT + (size_t)2048 * 1280 * 2;
static constexpr size_t OFF_SC  = OFF_HP  + (size_t)64 * 512 * 4;
static constexpr size_t OFF_AT  = OFF_SC  + (size_t)131072 * 4;
static constexpr size_t OFF_XIN = OFF_AT  + (size_t)131072 * 4;
static constexpr size_t OFF_Z   = OFF_XIN + (size_t)64 * 1280 * 4;
static constexpr size_t OFF_OUT = OFF_Z   + (size_t)2 * 64 * 2048 * 4;

extern "C" void kernel_launch(void* const* d_in, const int* in_sizes, int n_in,
                              void* d_out, int out_size, void* d_ws, size_t ws_size,
                              hipStream_t stream) {
  const int*   x     = (const int*)  d_in[0];
  const float* c_fwd = (const float*)d_in[1];
  const float* c_bwd = (const float*)d_in[2];
  const float* enc   = (const float*)d_in[3];
  const float* emb   = (const float*)d_in[4];
  const float* W1_w  = (const float*)d_in[5];
  const float* W1_b  = (const float*)d_in[6];
  const float* W2_w  = (const float*)d_in[7];
  const float* W2_b  = (const float*)d_in[8];
  const float* W3_w  = (const float*)d_in[9];
  const float* W3_b  = (const float*)d_in[10];
  const float* V_w   = (const float*)d_in[11];
  const float* V_b   = (const float*)d_in[12];
  const float* Kf    = (const float*)d_in[13];
  const float* bf_   = (const float*)d_in[14];
  const float* Kb    = (const float*)d_in[15];
  const float* bb_   = (const float*)d_in[16];
  const float* fc_w  = (const float*)d_in[17];
  const float* fc_b  = (const float*)d_in[18];

  char* ws = (char*)d_ws;
  unsigned short* W1t = (unsigned short*)(ws + OFF_W1T);
  unsigned short* KfT = (unsigned short*)(ws + OFF_KFT);
  unsigned short* KbT = (unsigned short*)(ws + OFF_KBT);
  float* hp    = (float*)(ws + OFF_HP);
  float* score = (float*)(ws + OFF_SC);
  float* attn  = (float*)(ws + OFF_AT);
  float* xin   = (float*)(ws + OFF_XIN);
  float* zbuf  = (float*)(ws + OFF_Z);
  float* outb  = (float*)(ws + OFF_OUT);
  float* out   = (float*)d_out;

  // 1. weight packing (fp32 -> bf16, transposed for B-fragment loads)
  pack_w1_kernel<<<(512 * 1024) / 256, 256, 0, stream>>>(W1_w, W1t);
  pack_k_kernel<<<dim3((2048 * 1280) / 256, 2), 256, 0, stream>>>(Kf, Kb, KfT, KbT);

  // 2. hidden projection (64x512)
  hproj_kernel<<<2, 256, 0, stream>>>(c_fwd, c_bwd, W2_w, W2_b, W3_w, W3_b, hp);

  // 3. fused attention scores: TDM-fed WMMA GEMM + tanh + V_w dot
  score_kernel<<<131072 / 64, 256, 0, stream>>>(enc, W1t, W1_b, hp, V_w, V_b, score);

  // 4. softmax over S, then context (streams enc once more)
  softmax_kernel<<<64, 256, 0, stream>>>(score, attn);
  ctx_kernel<<<dim3(64, 4), 256, 0, stream>>>(attn, enc, xin);
  emb_kernel<<<64, 256, 0, stream>>>(x, emb, xin);

  // 5. bidirectional LSTM gate GEMMs (WMMA)
  lstm_gemm_kernel<<<dim3(2048 / 64, 2), 256, 0, stream>>>(xin, KfT, KbT, bf_, bb_, zbuf);
  gates_kernel<<<dim3(128, 2), 256, 0, stream>>>(zbuf, outb, out);

  // 6. vocab projection: fc_w (205 MB) streamed exactly once
  logits_kernel<<<(50000 + 255) / 256, 256, 0, stream>>>(outb, fc_w, fc_b, out);
}
